// SelfAttention_85478439125486
// MI455X (gfx1250) — matrix operand order repaired and checked by reading, not compiled, not run on hardware
//
#include <hip/hip_runtime.h>
#include <hip/hip_bf16.h>
#include <math.h>

// Problem constants (match reference)
#define D_MODEL 512
#define N_HEADS 8
#define D_HEAD  64
#define SEQ     2048
#define BATCH   4
#define BS      (BATCH * SEQ)   // 8192 rows
#define LN_EPS  1e-5f

typedef _Float16 v16h __attribute__((ext_vector_type(16)));
typedef _Float16 h8   __attribute__((ext_vector_type(8)));
typedef float    v8f  __attribute__((ext_vector_type(8)));

__device__ __forceinline__ v8f wmma_f16(v16h a, v16h b, v8f c) {
    // (neg_a, A, neg_b, B, c_mod, C, reuse_a, reuse_b)
    return __builtin_amdgcn_wmma_f32_16x16x32_f16(false, a, false, b, (short)0, c, false, false);
}

// WMMA(f16) -> VALU RAW hazard guard (ISA 7.12.1: 4 co-exec NOPs).  The
// compiler's hazard recognizer does not see reads inside asm blocks, so the
// D-matrix value is threaded through 4 v_nops before any asm consumer.
__device__ __forceinline__ v8f xdl_guard(v8f x) {
    asm volatile("v_nop\n\tv_nop\n\tv_nop\n\tv_nop" : "+v"(x));
    return x;
}

// ---- CDNA5 async global->LDS copy (ASYNCcnt path), per ISA 15.18.3 --------
__device__ __forceinline__ unsigned lds_off(const void* p) {
    // flat LDS address: addr[31:0] is the LDS offset (ISA 10.2 aperture calc)
    return (unsigned)(size_t)p;
}
__device__ __forceinline__ void async_load_b128(unsigned ldsaddr, const void* gptr) {
    asm volatile("global_load_async_to_lds_b128 %0, %1, off"
                 :: "v"(ldsaddr), "v"((unsigned long long)(size_t)gptr)
                 : "memory");
}
__device__ __forceinline__ void wait_async0() {
    asm volatile("s_wait_asynccnt 0x0" ::: "memory");
}

// ---- raw v_max_num_f32 (no canonicalize, unlike fmaxf->llvm.maxnum) -------
__device__ __forceinline__ float vmax(float a, float b) {
    float d;
    asm("v_max_num_f32 %0, %1, %2" : "=v"(d) : "v"(a), "v"(b));
    return d;
}

// ---- 16-lane butterfly max: 4 instructions, DPP fused into the max --------
// xor1/2/4 via DPP8 selects, xor8-within-16 via DPP16 row_ror:8.
__device__ __forceinline__ float red_max16(float v) {
    float t;
    asm("v_max_num_f32_dpp %0, %1, %1 dpp8:[1,0,3,2,5,4,7,6]"
        : "=v"(t) : "v"(v));
    asm("v_max_num_f32_dpp %0, %1, %1 dpp8:[2,3,0,1,6,7,4,5]"
        : "=v"(v) : "v"(t));
    asm("v_max_num_f32_dpp %0, %1, %1 dpp8:[4,5,6,7,0,1,2,3]"
        : "=v"(t) : "v"(v));
    asm("v_max_num_f32_dpp %0, %1, %1 row_ror:8 row_mask:0xf bank_mask:0xf bound_ctrl:1"
        : "=v"(v) : "v"(t));
    return v;
}

// ---------------------------------------------------------------------------
// Kernel 1: weight transpose + f32 -> f16 convert.  wt[n][k] = (f16) w[k][n]
// ---------------------------------------------------------------------------
__global__ __launch_bounds__(256) void wtrans_kernel(const float* __restrict__ w,
                                                     _Float16* __restrict__ wt) {
    int idx = blockIdx.x * 256 + threadIdx.x;      // 0 .. 512*512-1
    int n = idx >> 9;
    int k = idx & 511;
    wt[(size_t)n * D_MODEL + k] = (_Float16)w[(size_t)k * D_MODEL + n];
}

// ---------------------------------------------------------------------------
// Kernel 2: LayerNorm.  One wave32 per 512-element row; 16 elems per lane.
// ---------------------------------------------------------------------------
__global__ __launch_bounds__(256) void ln_kernel(const float* __restrict__ x,
                                                 const float* __restrict__ gamma,
                                                 const float* __restrict__ beta,
                                                 _Float16* __restrict__ xn) {
    int wave = threadIdx.x >> 5;
    int lane = threadIdx.x & 31;
    int row  = blockIdx.x * 8 + wave;              // 8192 rows, grid.x = 1024

    const float* xr = x + (size_t)row * D_MODEL + lane * 16;
    float v[16];
#pragma unroll
    for (int i = 0; i < 4; ++i) {
        float4 t = ((const float4*)xr)[i];
        v[4 * i + 0] = t.x; v[4 * i + 1] = t.y;
        v[4 * i + 2] = t.z; v[4 * i + 3] = t.w;
    }
    float s = 0.f, ss = 0.f;
#pragma unroll
    for (int i = 0; i < 16; ++i) { s += v[i]; ss += v[i] * v[i]; }
#pragma unroll
    for (int off = 16; off > 0; off >>= 1) {
        s  += __shfl_xor(s, off, 32);
        ss += __shfl_xor(ss, off, 32);
    }
    float mean = s * (1.f / D_MODEL);
    float var  = ss * (1.f / D_MODEL) - mean * mean;
    float rstd = rsqrtf(var + LN_EPS);

    const float* gm = gamma + lane * 16;
    const float* bt = beta  + lane * 16;
    h8 o0, o1;
#pragma unroll
    for (int i = 0; i < 8; ++i)
        o0[i] = (_Float16)((v[i] - mean) * rstd * gm[i] + bt[i]);
#pragma unroll
    for (int i = 0; i < 8; ++i)
        o1[i] = (_Float16)((v[8 + i] - mean) * rstd * gm[8 + i] + bt[8 + i]);

    _Float16* orow = xn + (size_t)row * D_MODEL + lane * 16;
    *(h8*)orow       = o0;
    *((h8*)orow + 1) = o1;
}

// ---------------------------------------------------------------------------
// Kernel 3/5: WMMA GEMM.  C[8192x512] = A[8192x512] * Bt^T + bias (+residual)
// 16x512 A tile staged once per block via async global->LDS.  B fragments
// consumed immediately by their WMMA; k-loop unrolled x4.
// MODE: 0/1 = store f16 row-major (Q / K),
//       2   = store f16 transposed per head -> Vt (B,H,D_HEAD,SEQ),
//       3   = store f32 with bias + residual (output projection).
// ---------------------------------------------------------------------------
template <int MODE>
__global__ __launch_bounds__(128) void gemm_kernel(const _Float16* __restrict__ A,
                                                   const _Float16* __restrict__ Bt,
                                                   const float* __restrict__ bias,
                                                   const float* __restrict__ resid,
                                                   _Float16* __restrict__ out16,
                                                   float* __restrict__ out32) {
    __shared__ _Float16 atile[16 * D_MODEL];       // 16 KB
    int tid  = threadIdx.x;
    int wave = tid >> 5;
    int lane = tid & 31;
    int L = lane & 15, g = lane >> 4;
    int m0 = blockIdx.x * 16;                      // grid.x = 512
    int n0 = (blockIdx.y * 4 + wave) * 64;         // grid.y = 2

    // Stage A tile: 1024 x 16B chunks, 8 per thread.
#pragma unroll
    for (int it = 0; it < 8; ++it) {
        int chunk = it * 128 + tid;
        int row = chunk >> 6;                      // 64 chunks per 512-half row
        int col = (chunk & 63) * 8;
        async_load_b128(lds_off(&atile[row * D_MODEL + col]),
                        A + (size_t)(m0 + row) * D_MODEL + col);
    }
    wait_async0();
    __syncthreads();

    v8f acc[4] = {v8f{}, v8f{}, v8f{}, v8f{}};
    const _Float16* aL = atile + (size_t)L * D_MODEL;

#pragma unroll 4
    for (int k = 0; k < D_MODEL; k += 32) {
        v16h a;
        *(h8*)&a       = *(const h8*)(aL + k + g * 8);
        *((h8*)&a + 1) = *(const h8*)(aL + k + 16 + g * 8);
#pragma unroll
        for (int nf = 0; nf < 4; ++nf) {
            const _Float16* brow =
                Bt + (size_t)(n0 + nf * 16 + L) * D_MODEL + k + g * 8;
            __builtin_prefetch(brow + 32, 0, 1);    // global_prefetch_b8
            v16h b;
            *(h8*)&b       = *(const h8*)brow;
            *((h8*)&b + 1) = *(const h8*)(brow + 16);
            acc[nf] = wmma_f16(a, b, acc[nf]);
        }
    }

#pragma unroll
    for (int nf = 0; nf < 4; ++nf) {
#pragma unroll
        for (int r = 0; r < 8; ++r) {
            int m = m0 + r + g * 8;                // C/D layout: M = r + 8*(lane/16)
            int n = n0 + nf * 16 + L;              // N = lane%16 within tile
            float val = acc[nf][r] + bias[n];
            if (MODE == 0 || MODE == 1) {
                out16[(size_t)m * D_MODEL + n] = (_Float16)val;
            } else if (MODE == 2) {
                int b  = m >> 11, sr = m & (SEQ - 1);
                int h  = n >> 6,  d  = n & (D_HEAD - 1);
                out16[(((size_t)(b * N_HEADS + h) * D_HEAD + d) << 11) + sr] =
                    (_Float16)val;
            } else {
                out32[(size_t)m * D_MODEL + n] =
                    val + resid[(size_t)m * D_MODEL + n];
            }
        }
    }
}

// ---------------------------------------------------------------------------
// Kernel 4: flash attention.  Block = 4 waves = 64 queries of one (b,h).
// 64-key K/Vt tiles double-buffered in LDS via async global->LDS copies.
// Two 32-key sub-blocks per iteration; K fragment DS loads pipeline under
// the previous WMMA's execution.  Row max via 4-instruction DPP butterfly,
// row sum via one P*ones WMMA.  Softmax scale folded into Q fragments.
// ---------------------------------------------------------------------------
__global__ __launch_bounds__(128) void flash_kernel(const _Float16* __restrict__ Q,
                                                    const _Float16* __restrict__ K,
                                                    const _Float16* __restrict__ Vt,
                                                    _Float16* __restrict__ O) {
    __shared__ _Float16 kbuf[2][64 * 64];          // 8 KB per stage (keys x d)
    __shared__ _Float16 vbuf[2][64 * 64];          // 8 KB per stage (d x keys)
    __shared__ _Float16 pbufs[4][16 * 32];         // 1 KB per wave
    int tid  = threadIdx.x;
    int wave = tid >> 5;
    int lane = tid & 31;
    int L = lane & 15, g = lane >> 4;

    int bh = blockIdx.x;                           // 0..31
    int b  = bh >> 3, h = bh & 7;
    int q0 = (blockIdx.y * 4 + wave) * 16;         // grid.y = 32 -> q-tiles 0..127

    // Q A-fragments for d in [0,32) and [32,64); fold softmax scale into Q
    const float scale2 = 0.125f * 1.44269504f;     // 1/sqrt(d) * log2(e)
    const _Float16* qrow =
        Q + (size_t)(b * SEQ + q0 + L) * D_MODEL + h * D_HEAD;
    v16h qa0, qa1;
    *(h8*)&qa0       = *(const h8*)(qrow + g * 8);
    *((h8*)&qa0 + 1) = *(const h8*)(qrow + 16 + g * 8);
    *(h8*)&qa1       = *(const h8*)(qrow + 32 + g * 8);
    *((h8*)&qa1 + 1) = *(const h8*)(qrow + 48 + g * 8);
#pragma unroll
    for (int i = 0; i < 16; ++i) {
        qa0[i] = (_Float16)((float)qa0[i] * scale2);
        qa1[i] = (_Float16)((float)qa1[i] * scale2);
    }

    float mrow[8], lrow[8];
#pragma unroll
    for (int r = 0; r < 8; ++r) { mrow[r] = -3.0e38f; lrow[r] = 0.f; }
    v8f o[4] = {v8f{}, v8f{}, v8f{}, v8f{}};

    const _Float16* Kbh = K + (size_t)b * SEQ * D_MODEL + h * D_HEAD;
    const _Float16* Vbh = Vt + (size_t)bh * D_HEAD * SEQ;
    _Float16* pbuf = pbufs[wave];

    v16h ones;
#pragma unroll
    for (int i = 0; i < 16; ++i) ones[i] = (_Float16)1.0f;

    // cooperative staging: both tiles are 64 rows x 64 halves (128 B rows)
    int srow = tid >> 3;                           // 0..15
    int scol = (tid & 7) * 8;                      // halves

    auto stage = [&](int p, int kb) {
#pragma unroll
        for (int i = 0; i < 4; ++i) {
            int r = srow + i * 16;
            async_load_b128(lds_off(&kbuf[p][r * 64 + scol]),
                            Kbh + (size_t)(kb + r) * D_MODEL + scol);
            async_load_b128(lds_off(&vbuf[p][r * 64 + scol]),
                            Vbh + (size_t)r * SEQ + kb + scol);
        }
    };

    // load one 32-key sub-block's K fragments (4x v16h) from LDS
    auto load_kfrags = [&](int p, int koff, v16h* kf) {
#pragma unroll
        for (int nt = 0; nt < 2; ++nt) {
            const _Float16* krw = &kbuf[p][(koff + nt * 16 + L) * 64];
            *(h8*)&kf[nt * 2 + 0]       = *(const h8*)(krw + g * 8);
            *((h8*)&kf[nt * 2 + 0] + 1) = *(const h8*)(krw + 16 + g * 8);
            *(h8*)&kf[nt * 2 + 1]       = *(const h8*)(krw + 32 + g * 8);
            *((h8*)&kf[nt * 2 + 1] + 1) = *(const h8*)(krw + 48 + g * 8);
        }
    };

    // online softmax + PV for one 32-key sub-block given its score fragments
    auto process = [&](int p, int koff, v8f sc0, v8f sc1) {
#pragma unroll
        for (int r = 0; r < 8; ++r) {
            float s0 = sc0[r];
            float s1 = sc1[r];
            float mx = red_max16(vmax(s0, s1));    // DPP butterfly, no LDS
            float mnew  = vmax(mrow[r], mx);
            float alpha = exp2f(mrow[r] - mnew);
            float p0 = exp2f(s0 - mnew);
            float p1 = exp2f(s1 - mnew);
            lrow[r] *= alpha;
            mrow[r] = mnew;
            o[0][r] *= alpha; o[1][r] *= alpha;
            o[2][r] *= alpha; o[3][r] *= alpha;
            int m = r + g * 8;
            pbuf[m * 32 + L]      = (_Float16)p0;
            pbuf[m * 32 + 16 + L] = (_Float16)p1;
        }
        v16h pa;                                   // P in A-fragment layout
        *(h8*)&pa       = *(const h8*)(pbuf + L * 32 + g * 8);
        *((h8*)&pa + 1) = *(const h8*)(pbuf + L * 32 + 16 + g * 8);
        {
            v8f ls = {};
            ls = wmma_f16(pa, ones, ls);           // row sums of P
#pragma unroll
            for (int r = 0; r < 8; ++r) lrow[r] += ls[r];
        }
#pragma unroll
        for (int dt = 0; dt < 4; ++dt) {
            const _Float16* vrw = &vbuf[p][(dt * 16 + L) * 64 + koff + g * 8];
            v16h vb;
            *(h8*)&vb       = *(const h8*)vrw;
            *((h8*)&vb + 1) = *(const h8*)(vrw + 16);
            o[dt] = wmma_f16(pa, vb, o[dt]);
        }
    };

    stage(0, 0);
    wait_async0();
    __syncthreads();

    int p = 0;
    for (int kb = 0; kb < SEQ; kb += 64) {
        if (kb + 64 < SEQ) stage(p ^ 1, kb + 64);  // overlap with compute

        // K fragments for BOTH sub-blocks issued up front
        v16h kf0[4], kf1[4];
        load_kfrags(p, 0, kf0);
        load_kfrags(p, 32, kf1);

        // scores sub-block 0
        v8f a0 = {}, a1 = {};
        a0 = wmma_f16(qa0, kf0[0], a0);
        a1 = wmma_f16(qa0, kf0[2], a1);
        a0 = wmma_f16(qa1, kf0[1], a0);
        a1 = wmma_f16(qa1, kf0[3], a1);
        process(p, 0, xdl_guard(a0), xdl_guard(a1));  // hazard-safe reads

        // scores sub-block 1
        v8f b0 = {}, b1 = {};
        b0 = wmma_f16(qa0, kf1[0], b0);
        b1 = wmma_f16(qa0, kf1[2], b1);
        b0 = wmma_f16(qa1, kf1[1], b0);
        b1 = wmma_f16(qa1, kf1[3], b1);
        process(p, 32, xdl_guard(b0), xdl_guard(b1));

        wait_async0();            // staged next tile fully in LDS
        __syncthreads();          // all waves done reading buffer p
        p ^= 1;
    }

    // ---- normalize and store f16 (row-major B,S,D with head offset) ----
#pragma unroll
    for (int r = 0; r < 8; ++r) {
        float inv = 1.f / lrow[r];
        size_t row = (size_t)(b * SEQ + q0 + r + g * 8) * D_MODEL + h * D_HEAD;
#pragma unroll
        for (int dt = 0; dt < 4; ++dt)
            O[row + dt * 16 + L] = (_Float16)(o[dt][r] * inv);
    }
}

// ---------------------------------------------------------------------------
// Host-side launch
// ---------------------------------------------------------------------------
extern "C" void kernel_launch(void* const* d_in, const int* in_sizes, int n_in,
                              void* d_out, int out_size, void* d_ws, size_t ws_size,
                              hipStream_t stream) {
    const float* x     = (const float*)d_in[0];
    const float* wq    = (const float*)d_in[1];
    const float* bq    = (const float*)d_in[2];
    const float* wk    = (const float*)d_in[3];
    const float* bk    = (const float*)d_in[4];
    const float* wv    = (const float*)d_in[5];
    const float* bv    = (const float*)d_in[6];
    const float* wo    = (const float*)d_in[7];
    const float* bo    = (const float*)d_in[8];
    const float* gamma = (const float*)d_in[9];
    const float* beta  = (const float*)d_in[10];
    float* out = (float*)d_out;

    char* ws = (char*)d_ws;
    const size_t MB = 1u << 20;
    _Float16* xn16 = (_Float16*)(ws + 0 * MB);   // 8 MB
    _Float16* q16  = (_Float16*)(ws + 8 * MB);   // 8 MB
    _Float16* k16  = (_Float16*)(ws + 16 * MB);  // 8 MB
    _Float16* vt16 = (_Float16*)(ws + 24 * MB);  // 8 MB  (B,H,D_HEAD,SEQ)
    _Float16* at16 = (_Float16*)(ws + 32 * MB);  // 8 MB
    _Float16* wqt  = (_Float16*)(ws + 40 * MB);  // 0.5 MB each
    _Float16* wkt  = wqt + (size_t)D_MODEL * D_MODEL;
    _Float16* wvt  = wkt + (size_t)D_MODEL * D_MODEL;
    _Float16* wot  = wvt + (size_t)D_MODEL * D_MODEL;

    // 1. weight transpose + convert
    dim3 wtb(256), wtg((D_MODEL * D_MODEL) / 256);
    wtrans_kernel<<<wtg, wtb, 0, stream>>>(wq, wqt);
    wtrans_kernel<<<wtg, wtb, 0, stream>>>(wk, wkt);
    wtrans_kernel<<<wtg, wtb, 0, stream>>>(wv, wvt);
    wtrans_kernel<<<wtg, wtb, 0, stream>>>(wo, wot);

    // 2. LayerNorm -> xn f16
    ln_kernel<<<dim3(BS / 8), dim3(256), 0, stream>>>(x, gamma, beta, xn16);

    // 3. QKV projections (WMMA GEMMs)
    dim3 gg(BS / 16, 2), gb(128);
    gemm_kernel<0><<<gg, gb, 0, stream>>>(xn16, wqt, bq, nullptr, q16, nullptr);
    gemm_kernel<1><<<gg, gb, 0, stream>>>(xn16, wkt, bk, nullptr, k16, nullptr);
    gemm_kernel<2><<<gg, gb, 0, stream>>>(xn16, wvt, bv, nullptr, vt16, nullptr);

    // 4. flash attention
    flash_kernel<<<dim3(BATCH * N_HEADS, 32), dim3(128), 0, stream>>>(q16, k16, vt16, at16);

    // 5. output projection + bias + residual -> f32 out
    gemm_kernel<3><<<gg, gb, 0, stream>>>(at16, wot, bo, x, nullptr, out);
}
